// DAANNet_66778151518223
// MI455X (gfx1250) — matrix operand order repaired and checked by reading, not compiled, not run on hardware
//
#include <hip/hip_runtime.h>
#include <math.h>
#include <stdint.h>

// ---------------------------------------------------------------- constants
#define NN      10000     // nodes per graph
#define NE      320000    // edges per graph
#define IN_DIM  15962
#define D1      512
#define D2      64
#define BOTD    32
#define DHD     512

// ---------------------------------------------------------------- WMMA types
typedef __attribute__((ext_vector_type(16))) __bf16 bf16x16;
typedef __attribute__((ext_vector_type(8)))  float  f32x8;

union Frag  { uint4 u[2]; bf16x16 v; };
union Pack8 { __bf16 h[16]; uint4 u[2]; };   // native RNE converts, packed LDS stores

// ---------------------------------------------------------------- GEMM
// C[m, col0+n] = act( sum_k A[m,k]*B[k,n] + bias[n] )
// A:[M,K] f32 row-major, B:[K,N] f32 row-major, converted to bf16 (native
// v_cvt RNE) while staging into LDS; f32 accumulation via
// v_wmma_f32_16x16x32_bf16. Out-of-range rows/cols are CLAMPED (never
// zeroed): a garbage A-row/B-col only pollutes C rows/cols the epilogue
// masks anyway. Only the K tail needs zero-fill, handled by one peeled
// iteration with uniform control flow.
#define BM 128
#define BN 128
#define BK 32
#define LP 40      // LDS row pitch (elements): 32 + 8 pad, rows stay 16B-aligned

__device__ __forceinline__ void wmma_step(const unsigned short sA[BM][LP],
                                          const unsigned short sBt[BN][LP],
                                          int wm, int wn, int half, int l15,
                                          f32x8 acc[2][4])
{
  // A 16x32 bf16 fragment: lanes 0-15 -> K 0..7/16..23, lanes 16-31 -> K 8..15/24..31
  Frag fa[2];
#pragma unroll
  for (int mi = 0; mi < 2; ++mi) {
    int r = wm * 32 + mi * 16 + l15;
    fa[mi].u[0] = *(const uint4*)&sA[r][half * 8];
    fa[mi].u[1] = *(const uint4*)&sA[r][half * 8 + 16];
  }
  // B 32x16 bf16 fragment: lanes 0-15 -> K 0..7/16..23, lanes 16-31 -> K 8..15/24..31 (col = lane&15)
  Frag fb[4];
#pragma unroll
  for (int ni = 0; ni < 4; ++ni) {
    int r = wn * 64 + ni * 16 + l15;
    fb[ni].u[0] = *(const uint4*)&sBt[r][half * 8];
    fb[ni].u[1] = *(const uint4*)&sBt[r][half * 8 + 16];
  }
#pragma unroll
  for (int mi = 0; mi < 2; ++mi)
#pragma unroll
    for (int ni = 0; ni < 4; ++ni)
      acc[mi][ni] = __builtin_amdgcn_wmma_f32_16x16x32_bf16(
          false, fa[mi].v, false, fb[ni].v, (short)0, acc[mi][ni], false, false);
}

__global__ __launch_bounds__(256)
void gemm_bf16_wmma(const float* __restrict__ A, const float* __restrict__ B,
                    const float* __restrict__ bias, float* __restrict__ C,
                    int M, int N, int K, int ldc, int col0, int act)
{
  __shared__ __align__(16) unsigned short sA[BM][LP];   // A tile  [m][k]
  __shared__ __align__(16) unsigned short sBt[BN][LP];  // B tile transposed [n][k]

  const int tid  = threadIdx.x;
  const int lane = tid & 31, wave = tid >> 5;
  const int wm   = wave & 3, wn = wave >> 2;      // 4x2 wave grid -> 128x128
  const int half = lane >> 4, l15 = lane & 15;
  const int m0 = blockIdx.y * BM, n0 = blockIdx.x * BN;

  f32x8 acc[2][4];
  const f32x8 z = {0.f,0.f,0.f,0.f,0.f,0.f,0.f,0.f};
#pragma unroll
  for (int mi = 0; mi < 2; ++mi)
#pragma unroll
    for (int ni = 0; ni < 4; ++ni) acc[mi][ni] = z;

  // A staging: thread -> (row, 16-wide k chunk); contiguous 64B per thread
  const int ar = tid >> 1, ac = (tid & 1) * 16;
  // B staging: thread -> (column, 16-wide k chunk); lane-coalesced global reads
  const int bn = tid & 127, bkc = (tid >> 7) * 16;

  const float* Arow = A + (long long)min(m0 + ar, M - 1) * K;  // clamped row
  const int    bcol = min(n0 + bn, N - 1);                     // clamped col

  const int KF = (K / BK) * BK;   // full-tile portion of K

  for (int k0 = 0; k0 < KF; k0 += BK) {
    __syncthreads();
    { // ---- stage A tile (f32 -> bf16), unguarded
      const float* Ap = Arow + k0 + ac;
      Pack8 pk;
#pragma unroll
      for (int j = 0; j < 16; ++j) pk.h[j] = (__bf16)Ap[j];
      *(uint4*)&sA[ar][ac]     = pk.u[0];
      *(uint4*)&sA[ar][ac + 8] = pk.u[1];
      if (k0 + BK < K) __builtin_prefetch(Ap + BK, 0, 0);
    }
    { // ---- stage B tile transposed (f32 -> bf16), unguarded
      const float* Bp = B + (long long)(k0 + bkc) * N + bcol;
      Pack8 pk;
#pragma unroll
      for (int j = 0; j < 16; ++j) pk.h[j] = (__bf16)Bp[(long long)j * N];
      *(uint4*)&sBt[bn][bkc]     = pk.u[0];
      *(uint4*)&sBt[bn][bkc + 8] = pk.u[1];
      if (k0 + BK < K) __builtin_prefetch(Bp + (long long)BK * N, 0, 0);
    }
    __syncthreads();
    wmma_step(sA, sBt, wm, wn, half, l15, acc);
  }

  if (KF < K) {   // ---- peeled K tail: uniform flow, load clamped + select 0
    __syncthreads();
    {
      Pack8 pk;
#pragma unroll
      for (int j = 0; j < 16; ++j) {
        int k = KF + ac + j;
        float v = Arow[min(k, K - 1)];
        pk.h[j] = (__bf16)(k < K ? v : 0.f);
      }
      *(uint4*)&sA[ar][ac]     = pk.u[0];
      *(uint4*)&sA[ar][ac + 8] = pk.u[1];
    }
    {
      Pack8 pk;
#pragma unroll
      for (int j = 0; j < 16; ++j) {
        int k = KF + bkc + j;
        float v = B[(long long)min(k, K - 1) * N + bcol];
        pk.h[j] = (__bf16)(k < K ? v : 0.f);
      }
      *(uint4*)&sBt[bn][bkc]     = pk.u[0];
      *(uint4*)&sBt[bn][bkc + 8] = pk.u[1];
    }
    __syncthreads();
    wmma_step(sA, sBt, wm, wn, half, l15, acc);
  }

  // C/D layout: VGPR r, lanes 0-15: (M=r, N=lane); lanes 16-31: (M=r+8, N=lane-16)
  if ((m0 + BM <= M) && (n0 + BN <= N)) {
#pragma unroll
    for (int mi = 0; mi < 2; ++mi)
#pragma unroll
      for (int ni = 0; ni < 4; ++ni) {
        int gn = n0 + wn * 64 + ni * 16 + l15;
        float bv = bias ? bias[gn] : 0.f;
#pragma unroll
        for (int r = 0; r < 8; ++r) {
          int gm = m0 + wm * 32 + mi * 16 + half * 8 + r;
          float v = acc[mi][ni][r] + bv;
          if (act) v = fmaxf(v, 0.f);
          C[(long long)gm * ldc + col0 + gn] = v;
        }
      }
  } else {
#pragma unroll
    for (int mi = 0; mi < 2; ++mi)
#pragma unroll
      for (int ni = 0; ni < 4; ++ni)
#pragma unroll
        for (int r = 0; r < 8; ++r) {
          int gm = m0 + wm * 32 + mi * 16 + half * 8 + r;
          int gn = n0 + wn * 64 + ni * 16 + l15;
          if (gm < M && gn < N) {
            float v = acc[mi][ni][r];
            if (bias) v += bias[gn];
            if (act)  v = fmaxf(v, 0.f);
            C[(long long)gm * ldc + col0 + gn] = v;
          }
        }
  }
}

// ---------------------------------------------------------------- edge / elementwise
// monotone order-preserving float<->uint encoding for atomicMax on floats
__device__ __forceinline__ unsigned fenc(float f) {
  unsigned b = __float_as_uint(f);
  return (b & 0x80000000u) ? ~b : (b | 0x80000000u);
}
__device__ __forceinline__ float fdec(unsigned u) {
  unsigned b = (u & 0x80000000u) ? (u & 0x7FFFFFFFu) : ~u;
  return __uint_as_float(b);
}
#define ENC_NEGINF 0x007FFFFFu   // fenc(-inf)

__global__ void seg_init(unsigned* umax, float* ssum, int n) {
  int i = blockIdx.x * blockDim.x + threadIdx.x;
  if (i < n) { umax[i] = ENC_NEGINF; ssum[i] = 0.f; }
}

__global__ void seg_finalize(const unsigned* umax, float* m, int n) {
  int i = blockIdx.x * blockDim.x + threadIdx.x;
  if (i >= n) return;
  float v = fdec(umax[i]);
  // empty segments (-inf) -> 0, matching jnp.where(isfinite(m), m, 0)
  if (!(v >= -3.0e38f && v <= 3.0e38f)) v = 0.f;
  m[i] = v;
}

// one wave per edge: score = dot(q[dst], k[src]) * inv_sqrt_d; atomicMax per dst
__global__ void edge_scores(const int* __restrict__ ei, const float* __restrict__ H,
                            int ld, int qcol, int kcol, int d, float scale,
                            float* __restrict__ score, unsigned* __restrict__ umax, int E)
{
  int w = (blockIdx.x * blockDim.x + threadIdx.x) >> 5;
  int lane = threadIdx.x & 31;
  if (w >= E) return;
  int src = ei[w], dst = ei[E + w];
  const float* q = H + (long long)dst * ld + qcol;
  const float* k = H + (long long)src * ld + kcol;
  float s = 0.f;
  for (int i = lane; i < d; i += 32) s += q[i] * k[i];
#pragma unroll
  for (int m = 16; m; m >>= 1) s += __shfl_xor(s, m, 32);
  s *= scale;
  if (lane == 0) { score[w] = s; atomicMax(&umax[dst], fenc(s)); }
}

__global__ void edge_expsum(const int* __restrict__ ei, const float* __restrict__ score,
                            const float* __restrict__ m, float* __restrict__ eexp,
                            float* __restrict__ ssum, int E)
{
  int e = blockIdx.x * blockDim.x + threadIdx.x;
  if (e >= E) return;
  int dst = ei[E + e];
  float v = expf(score[e] - m[dst]);
  eexp[e] = v;
  atomicAdd(&ssum[dst], v);
}

// one wave per edge: agg[dst] += (e/(s+eps)) * v[src]
__global__ void edge_aggregate(const int* __restrict__ ei, const float* __restrict__ eexp,
                               const float* __restrict__ ssum, const float* __restrict__ H,
                               int ld, int vcol, float* __restrict__ agg, int d, int E)
{
  int w = (blockIdx.x * blockDim.x + threadIdx.x) >> 5;
  int lane = threadIdx.x & 31;
  if (w >= E) return;
  int src = ei[w], dst = ei[E + w];
  float a = eexp[w] / (ssum[dst] + 1e-16f);
  const float* v = H + (long long)src * ld + vcol;
  float* o = agg + (long long)dst * d;
  for (int i = lane; i < d; i += 32) atomicAdd(&o[i], a * v[i]);
}

// out[n*d + j] = in[n*ld + col0 + j]
__global__ void copy_cols(const float* __restrict__ in, float* __restrict__ out,
                          int n, int d, int ld, int col0)
{
  long long i = (long long)blockIdx.x * blockDim.x + threadIdx.x;
  if (i >= (long long)n * d) return;
  int row = (int)(i / d), j = (int)(i % d);
  out[i] = in[(long long)row * ld + col0 + j];
}

__global__ void elu_inplace(float* __restrict__ x, long long cnt) {
  long long i = (long long)blockIdx.x * blockDim.x + threadIdx.x;
  if (i >= cnt) return;
  float v = x[i];
  x[i] = v > 0.f ? v : expm1f(v);
}

__global__ void softmax2(const float* __restrict__ emb, float* __restrict__ p, int n) {
  int i = blockIdx.x * blockDim.x + threadIdx.x;
  if (i >= n) return;
  float e0 = emb[2 * i], e1 = emb[2 * i + 1];
  float mx = fmaxf(e0, e1);
  float a = expf(e0 - mx), b = expf(e1 - mx);
  float inv = 1.f / (a + b);
  p[2 * i] = a * inv; p[2 * i + 1] = b * inv;
}

// fc[n,:] = p[n,cls] * share[n,:]
__global__ void scale_rows(const float* __restrict__ p, const float* __restrict__ share,
                           float* __restrict__ fc, int cls, int n, int d)
{
  int i = blockIdx.x * blockDim.x + threadIdx.x;
  if (i >= n * d) return;
  int row = i / d;
  fc[i] = p[2 * row + cls] * share[i];
}

// ---------------------------------------------------------------- host
static inline dim3 gemm_grid(int M, int N) {
  return dim3((unsigned)((N + BN - 1) / BN), (unsigned)((M + BM - 1) / BM));
}

extern "C" void kernel_launch(void* const* d_in, const int* in_sizes, int n_in,
                              void* d_out_v, int out_size, void* d_ws, size_t ws_size,
                              hipStream_t stream)
{
  (void)in_sizes; (void)n_in; (void)out_size; (void)ws_size;

  const float* src_x  = (const float*)d_in[0];
  const float* tgt_x  = (const float*)d_in[1];
  const int*   src_ei = (const int*)d_in[2];
  const int*   tgt_ei = (const int*)d_in[3];
  const float *Wq1=(const float*)d_in[4],  *bq1=(const float*)d_in[5];
  const float *Wk1=(const float*)d_in[6],  *bk1=(const float*)d_in[7];
  const float *Wv1=(const float*)d_in[8],  *bv1=(const float*)d_in[9];
  const float *Ws1=(const float*)d_in[10], *bs1=(const float*)d_in[11];
  const float *Wq2=(const float*)d_in[12], *bq2=(const float*)d_in[13];
  const float *Wk2=(const float*)d_in[14], *bk2=(const float*)d_in[15];
  const float *Wv2=(const float*)d_in[16], *bv2=(const float*)d_in[17];
  const float *Ws2=(const float*)d_in[18], *bs2=(const float*)d_in[19];
  const float *Wb =(const float*)d_in[20], *bb =(const float*)d_in[21];
  const float *Wc =(const float*)d_in[22], *bc =(const float*)d_in[23];
  const float *Wd1=(const float*)d_in[24], *bd1=(const float*)d_in[25];
  const float *Wd2=(const float*)d_in[26], *bd2=(const float*)d_in[27];
  const float *Wd3=(const float*)d_in[28], *bd3=(const float*)d_in[29];
  const float *Wdc1=(const float*)d_in[30],*bdc1=(const float*)d_in[31];
  const float *Wdc2=(const float*)d_in[32],*bdc2=(const float*)d_in[33];
  const float *Wdc3=(const float*)d_in[34],*bdc3=(const float*)d_in[35];
  float* out = (float*)d_out_v;

  // workspace bump allocator (~165 MB total)
  char* base = (char*)d_ws; size_t off = 0;
  auto alloc = [&](size_t bytes) -> void* {
    off = (off + 255) & ~(size_t)255;
    void* p = base + off; off += bytes; return p;
  };
  float*    H1    = (float*)   alloc((size_t)NN * 2048 * 4);  // q|k|v|s layer1
  float*    H2    = (float*)   alloc((size_t)NN * 256  * 4);  // q|k|v|s layer2
  float*    agg1  = (float*)   alloc((size_t)NN * D1   * 4);
  float*    agg2  = (float*)   alloc((size_t)NN * D2   * 4);
  float*    share = (float*)   alloc((size_t)NN * BOTD * 4);
  float*    emb   = (float*)   alloc((size_t)NN * 2    * 4);
  float*    prob  = (float*)   alloc((size_t)NN * 2    * 4);
  float*    score = (float*)   alloc((size_t)NE * 4);
  float*    eexp  = (float*)   alloc((size_t)NE * 4);
  unsigned* umax  = (unsigned*)alloc((size_t)NN * 4);
  float*    mseg  = (float*)   alloc((size_t)NN * 4);
  float*    ssum  = (float*)   alloc((size_t)NN * 4);
  float*    hdA   = (float*)   alloc((size_t)NN * DHD * 4);
  float*    hdB   = (float*)   alloc((size_t)NN * DHD * 4);
  float*    fc    = (float*)   alloc((size_t)NN * BOTD * 4);

  const int T = 256;
  auto g1 = [&](long long n) { return (unsigned)((n + T - 1) / T); };
  const unsigned edgeBlocks = (NE + 7) / 8;   // 8 wave32 per block, 1 wave/edge

  const float* WL1[4] = {Wq1, Wk1, Wv1, Ws1}; const float* bL1[4] = {bq1, bk1, bv1, bs1};
  const float* WL2[4] = {Wq2, Wk2, Wv2, Ws2}; const float* bL2[4] = {bq2, bk2, bv2, bs2};

  auto run_graph = [&](const float* x, const int* ei,
                       float* emb_out, float* dom_out, float* cls_out)
  {
    // ---- layer 1 projections: [NN,IN] x [IN,512] -> H1[:, g*512..]
    for (int g = 0; g < 4; ++g)
      gemm_bf16_wmma<<<gemm_grid(NN, D1), T, 0, stream>>>(
          x, WL1[g], bL1[g], H1, NN, D1, IN_DIM, 2048, g * D1, 0);

    // ---- layer 1 scatter-softmax attention (q@0, k@512, v@1024, s@1536)
    seg_init<<<g1(NN), T, 0, stream>>>(umax, ssum, NN);
    edge_scores<<<edgeBlocks, T, 0, stream>>>(ei, H1, 2048, 0, D1, D1,
                                              1.0f / sqrtf((float)D1), score, umax, NE);
    seg_finalize<<<g1(NN), T, 0, stream>>>(umax, mseg, NN);
    edge_expsum<<<g1(NE), T, 0, stream>>>(ei, score, mseg, eexp, ssum, NE);
    copy_cols<<<g1((long long)NN * D1), T, 0, stream>>>(H1, agg1, NN, D1, 2048, 3 * D1);
    edge_aggregate<<<edgeBlocks, T, 0, stream>>>(ei, eexp, ssum, H1, 2048, 2 * D1, agg1, D1, NE);
    elu_inplace<<<g1((long long)NN * D1), T, 0, stream>>>(agg1, (long long)NN * D1);

    // ---- layer 2 projections: [NN,512] x [512,64] -> H2[:, g*64..]
    for (int g = 0; g < 4; ++g)
      gemm_bf16_wmma<<<gemm_grid(NN, D2), T, 0, stream>>>(
          agg1, WL2[g], bL2[g], H2, NN, D2, D1, 256, g * D2, 0);

    // ---- layer 2 attention (q@0, k@64, v@128, s@192)
    seg_init<<<g1(NN), T, 0, stream>>>(umax, ssum, NN);
    edge_scores<<<edgeBlocks, T, 0, stream>>>(ei, H2, 256, 0, D2, D2,
                                              1.0f / sqrtf((float)D2), score, umax, NE);
    seg_finalize<<<g1(NN), T, 0, stream>>>(umax, mseg, NN);
    edge_expsum<<<g1(NE), T, 0, stream>>>(ei, score, mseg, eexp, ssum, NE);
    copy_cols<<<g1((long long)NN * D2), T, 0, stream>>>(H2, agg2, NN, D2, 256, 3 * D2);
    edge_aggregate<<<edgeBlocks, T, 0, stream>>>(ei, eexp, ssum, H2, 256, 2 * D2, agg2, D2, NE);
    elu_inplace<<<g1((long long)NN * D2), T, 0, stream>>>(agg2, (long long)NN * D2);

    // ---- bottleneck + classifier + softmax
    gemm_bf16_wmma<<<gemm_grid(NN, BOTD), T, 0, stream>>>(agg2, Wb, bb, share,
                                                          NN, BOTD, D2, BOTD, 0, 0);
    gemm_bf16_wmma<<<gemm_grid(NN, 2), T, 0, stream>>>(share, Wc, bc, emb,
                                                       NN, 2, BOTD, 2, 0, 0);
    if (emb_out)
      copy_cols<<<g1((long long)NN * 2), T, 0, stream>>>(emb, emb_out, NN, 2, 2, 0);
    softmax2<<<g1(NN), T, 0, stream>>>(emb, prob, NN);

    // ---- domain head (grad_reverse is identity in forward): 32->512->512->2
    gemm_bf16_wmma<<<gemm_grid(NN, DHD), T, 0, stream>>>(share, Wd1, bd1, hdA,
                                                         NN, DHD, BOTD, DHD, 0, 1);
    gemm_bf16_wmma<<<gemm_grid(NN, DHD), T, 0, stream>>>(hdA, Wd2, bd2, hdB,
                                                         NN, DHD, DHD, DHD, 0, 1);
    gemm_bf16_wmma<<<gemm_grid(NN, 2), T, 0, stream>>>(hdB, Wd3, bd3, dom_out,
                                                       NN, 2, DHD, 2, 0, 0);

    // ---- per-class heads: f = p[:,c]*share -> 32->512->512->2
    for (int c = 0; c < 2; ++c) {
      scale_rows<<<g1((long long)NN * BOTD), T, 0, stream>>>(prob, share, fc, c, NN, BOTD);
      gemm_bf16_wmma<<<gemm_grid(NN, DHD), T, 0, stream>>>(
          fc, Wdc1 + (size_t)c * BOTD * DHD, bdc1 + (size_t)c * DHD, hdA,
          NN, DHD, BOTD, DHD, 0, 1);
      gemm_bf16_wmma<<<gemm_grid(NN, DHD), T, 0, stream>>>(
          hdA, Wdc2 + (size_t)c * DHD * DHD, bdc2 + (size_t)c * DHD, hdB,
          NN, DHD, DHD, DHD, 0, 1);
      gemm_bf16_wmma<<<gemm_grid(NN, 2), T, 0, stream>>>(
          hdB, Wdc3 + (size_t)c * DHD * 2, bdc3 + (size_t)c * 2,
          cls_out + (size_t)c * NN * 2, NN, 2, DHD, 2, 0, 0);
    }
  };

  // d_out layout: source_emb[0), s_dom[20000), t_dom[40000), s_out[60000), t_out[100000)
  run_graph(src_x, src_ei, out + 0,      out + 20000, out + 60000);
  run_graph(tgt_x, tgt_ei, (float*)0,    out + 40000, out + 100000);
}
